// GraphWaveletNeuralNetwork_27281632264453
// MI455X (gfx1250) — matrix operand order repaired and checked by reading, not compiled, not run on hardware
//
#include <hip/hip_runtime.h>

// GWNN on MI455X (gfx1250, wave32, WMMA).
// All heavy math runs through v_wmma_f32_16x16x32_f16 with f32 accumulation.
// diag(filter) is folded into a row-scale epilogue on the spectral matrix so
// the 256MB wavelet matrices are streamed unmodified (~1 GB total HBM, the
// bandwidth floor at 23.3 TB/s). B tiles are staged with the CDNA5 async
// global->LDS data mover (ASYNCcnt), double-buffered in LDS; B fragments are
// register-blocked so the 8 WMMAs issue back-to-back.

typedef __attribute__((ext_vector_type(16))) _Float16 v16h;
typedef __attribute__((ext_vector_type(8)))  _Float16 v8h;
typedef __attribute__((ext_vector_type(8)))  float    v8f;
typedef __attribute__((ext_vector_type(4)))  float    v4f;

#define NDIM   8192
#define NC     256
#define BM     64
#define BK     32
#define THREADS 256
#define LDS_STRIDE 40   // halves; 80B rows (16B aligned), pads bank conflicts

enum Epi { EPI_F16T = 0, EPI_F16T_SCALE = 1, EPI_F16_ROW_RELU = 2, EPI_F32_ROW = 3 };

// Stage 64B (this thread's tile row) global -> LDS via the async data mover.
// INST_OFFSET is added to BOTH the LDS and the global address (ISA ch.10.4).
__device__ __forceinline__ void async_stage64(unsigned lds_off,
                                              unsigned long long gaddr) {
    asm volatile(
        "global_load_async_to_lds_b128 %0, %1, off\n\t"
        "global_load_async_to_lds_b128 %0, %1, off offset:16\n\t"
        "global_load_async_to_lds_b128 %0, %1, off offset:32\n\t"
        "global_load_async_to_lds_b128 %0, %1, off offset:48"
        :: "v"(lds_off), "v"(gaddr)
        : "memory");
}

__device__ __forceinline__ void wait_async0() {
    asm volatile("s_wait_asynccnt 0x0" ::: "memory");
}

// C[M,256] = A[M,K] @ B[K,256], where B is passed TRANSPOSED as Bt[256,K] f16.
// A is f32 (converted to f16 in registers) or f16 (A_F16=true).
template <bool A_F16, int EPI>
__global__ __launch_bounds__(THREADS)
void gemm_wmma(const void* __restrict__ Av, const _Float16* __restrict__ Bt,
               void* __restrict__ Cv, const float* __restrict__ filt, int K)
{
    __shared__ _Float16 Bs[2][NC * LDS_STRIDE];   // double-buffered, 40 KB

    const int tid     = threadIdx.x;
    const int lane    = tid & 31;
    const int wave    = tid >> 5;     // 0..7
    const int mw      = wave & 3;     // wave's 16-row stripe within 64-row block
    const int nw      = wave >> 2;    // 0..1 -> columns [nw*128, nw*128+128)
    const int lane_lo = lane & 15;
    const int lane_hi = lane >> 4;    // 0 or 1
    const int khalf   = lane_hi * 8;  // ISA A-layout K-offset per lane half

    const int m_frag = blockIdx.x * BM + mw * 16;
    const int niter  = K / BK;

    // per-thread staging addresses (thread t owns Bt row t, 64B per K-step)
    const unsigned long long gB =
        (unsigned long long)(const void*)(Bt + (size_t)tid * K);
    const unsigned lds0 =
        (unsigned)(uintptr_t)(void*)&Bs[0][tid * LDS_STRIDE];
    const unsigned lds1 =
        (unsigned)(uintptr_t)(void*)&Bs[1][tid * LDS_STRIDE];

    const float*    gA32 = (const float*)Av
                           + (size_t)(m_frag + lane_lo) * K + khalf;
    const _Float16* gA16 = (const _Float16*)Av
                           + (size_t)(m_frag + lane_lo) * K + khalf;

    // A prefetch registers (loaded for step i+1 while computing step i)
    v4f ar0, ar1, ar2, ar3;   // A_F16 == false
    v8h ah0, ah1;             // A_F16 == true

    auto loadA = [&](int k0) {
        if (!A_F16) {
            const float* ap = gA32 + k0;
            ar0 = *(const v4f*)(ap);      ar1 = *(const v4f*)(ap + 4);
            ar2 = *(const v4f*)(ap + 16); ar3 = *(const v4f*)(ap + 20);
        } else {
            const _Float16* ap = gA16 + k0;
            ah0 = *(const v8h*)(ap); ah1 = *(const v8h*)(ap + 16);
        }
    };

    // ---- prologue: async-stage tile 0, prefetch A fragment 0
    async_stage64(lds0, gB);
    loadA(0);

    v8f acc[8] = {};

    for (int i = 0; i < niter; ++i) {
        // drain this wave's async stage from the previous step, then sync.
        // After the barrier: tile i is fully in Bs[i&1] (all waves drained
        // before signaling) and every wave is done reading Bs[(i+1)&1].
        wait_async0();
        __syncthreads();

        // immediately kick off the async copy of tile i+1 into the buffer
        // that just became free; it has the whole compute phase to land.
        if (i + 1 < niter)
            async_stage64((i + 1) & 1 ? lds1 : lds0,
                          gB + (unsigned long long)((i + 1) * BK) * 2u);

        // convert current A fragment (16x32, ISA 7.12.2 layout)
        v16h afrag;
        if (!A_F16) {
            #pragma unroll
            for (int h = 0; h < 4; ++h) {
                afrag[h]      = (_Float16)ar0[h];
                afrag[h + 4]  = (_Float16)ar1[h];
                afrag[h + 8]  = (_Float16)ar2[h];
                afrag[h + 12] = (_Float16)ar3[h];
            }
        } else {
            #pragma unroll
            for (int h = 0; h < 8; ++h) { afrag[h] = ah0[h]; afrag[h + 8] = ah1[h]; }
        }

        // prefetch next step's A fragment during compute
        if (i + 1 < niter) loadA((i + 1) * BK);
        // deep prefetch of the HBM A stream
        if (i + 8 < niter) {
            if (!A_F16) __builtin_prefetch(gA32 + (i + 8) * BK, 0, 0);
            else        __builtin_prefetch(gA16 + (i + 8) * BK, 0, 0);
        }

        // register-block all 8 B fragments, then 8 back-to-back WMMAs
        const _Float16* base = &Bs[i & 1][0];
        v16h bf[8];
        #pragma unroll
        for (int nt = 0; nt < 8; ++nt) {
            const int n_local = nw * 128 + nt * 16 + lane_lo;
            const _Float16* bp = base + n_local * LDS_STRIDE + lane_hi * 8;
            v8h b0 = *(const v8h*)(bp);
            v8h b1 = *(const v8h*)(bp + 16);
            #pragma unroll
            for (int h = 0; h < 8; ++h) { bf[nt][h] = b0[h]; bf[nt][h + 8] = b1[h]; }
        }
        #pragma unroll
        for (int nt = 0; nt < 8; ++nt) {
            acc[nt] = __builtin_amdgcn_wmma_f32_16x16x32_f16(
                false, afrag, false, bf[nt], (short)0, acc[nt], false, false);
        }
    }

    // ---- epilogue. C tile layout: VGPR r -> M = r + 8*lane_hi, N = lane_lo
    const int m0 = m_frag + lane_hi * 8;
    v4f s0, s1;
    if (EPI == EPI_F16T_SCALE) {    // filter scale: 8 consecutive m per lane
        s0 = *(const v4f*)(filt + m0);
        s1 = *(const v4f*)(filt + m0 + 4);
    }
    #pragma unroll
    for (int nt = 0; nt < 8; ++nt) {
        const int n_out = nw * 128 + nt * 16 + lane_lo;
        if (EPI == EPI_F32_ROW) {
            #pragma unroll
            for (int r = 0; r < 8; ++r)
                ((float*)Cv)[(size_t)(m0 + r) * NC + n_out] = acc[nt][r];
        } else if (EPI == EPI_F16_ROW_RELU) {
            #pragma unroll
            for (int r = 0; r < 8; ++r)
                ((_Float16*)Cv)[(size_t)(m0 + r) * NC + n_out] =
                    (_Float16)fmaxf(acc[nt][r], 0.f);
        } else {
            // transposed f16 store: 8 consecutive m -> one 16B packed store
            v8h pk;
            if (EPI == EPI_F16T_SCALE) {
                #pragma unroll
                for (int r = 0; r < 4; ++r) {
                    pk[r]     = (_Float16)(acc[nt][r]     * s0[r]);
                    pk[r + 4] = (_Float16)(acc[nt][r + 4] * s1[r]);
                }
            } else {
                #pragma unroll
                for (int r = 0; r < 8; ++r) pk[r] = (_Float16)acc[nt][r];
            }
            *(v8h*)((_Float16*)Cv + (size_t)n_out * NDIM + m0) = pk;
        }
    }
}

// W [K,Ncol] f32 -> Wt [Ncol,K] f16 (tiny weight matrices only)
__global__ void transpose_to_f16(const float* __restrict__ W,
                                 _Float16* __restrict__ Wt, int K, int Ncol)
{
    int idx = blockIdx.x * blockDim.x + threadIdx.x;
    if (idx < K * Ncol) {
        int k = idx / Ncol, n = idx % Ncol;
        Wt[(size_t)n * K + k] = (_Float16)W[(size_t)k * Ncol + n];
    }
}

extern "C" void kernel_launch(void* const* d_in, const int* in_sizes, int n_in,
                              void* d_out, int out_size, void* d_ws, size_t ws_size,
                              hipStream_t stream) {
    const float* input = (const float*)d_in[0];   // [8192, 512]
    const float* wav   = (const float*)d_in[1];   // [8192, 8192]
    const float* wavi  = (const float*)d_in[2];   // [8192, 8192]
    const float* W1    = (const float*)d_in[3];   // [512, 256]
    const float* W2    = (const float*)d_in[4];   // [256, 256]
    const float* f1    = (const float*)d_in[5];   // [8192]
    const float* f2    = (const float*)d_in[6];   // [8192]

    char* ws = (char*)d_ws;
    _Float16* W1t = (_Float16*)ws; ws += (size_t)NC * 512 * sizeof(_Float16);
    _Float16* W2t = (_Float16*)ws; ws += (size_t)NC * NC  * sizeof(_Float16);
    _Float16* T0  = (_Float16*)ws; ws += (size_t)NC * NDIM * sizeof(_Float16);
    _Float16* T1  = (_Float16*)ws; ws += (size_t)NC * NDIM * sizeof(_Float16);
    _Float16* H1  = (_Float16*)ws; ws += (size_t)NDIM * NC * sizeof(_Float16);

    dim3 blk(THREADS);
    transpose_to_f16<<<(512 * NC + 255) / 256, blk, 0, stream>>>(W1, W1t, 512, NC);
    transpose_to_f16<<<(NC * NC + 255) / 256, blk, 0, stream>>>(W2, W2t, NC, NC);

    dim3 grid(NDIM / BM);   // 128 blocks, each owns a disjoint 64-row slab of A

    // layer 1
    gemm_wmma<false, EPI_F16T>        <<<grid, blk, 0, stream>>>(input, W1t, T0, nullptr, 512);  // t1^T
    gemm_wmma<false, EPI_F16T_SCALE>  <<<grid, blk, 0, stream>>>(wavi,  T0,  T1, f1, NDIM);      // (f1⊙s1)^T
    gemm_wmma<false, EPI_F16_ROW_RELU><<<grid, blk, 0, stream>>>(wav,   T1,  H1, nullptr, NDIM); // h1
    // layer 2
    gemm_wmma<true,  EPI_F16T>        <<<grid, blk, 0, stream>>>(H1,  W2t, T0, nullptr, NC);     // t2^T
    gemm_wmma<false, EPI_F16T_SCALE>  <<<grid, blk, 0, stream>>>(wavi, T0, T1, f2, NDIM);        // (f2⊙s2)^T
    gemm_wmma<false, EPI_F32_ROW>     <<<grid, blk, 0, stream>>>(wav,  T1, d_out, nullptr, NDIM);// out
}
